// LocalTransformer_31086973288642
// MI455X (gfx1250) — matrix operand order repaired and checked by reading, not compiled, not run on hardware
//
#include <hip/hip_runtime.h>
#include <cstdint>
#include <cstddef>

// ---------------------------------------------------------------------------
// Types / WMMA helpers (gfx1250 wave32, V_WMMA_F32_16X16X32_F16)
// ---------------------------------------------------------------------------
typedef _Float16 half8   __attribute__((ext_vector_type(8)));
typedef _Float16 half16  __attribute__((ext_vector_type(16)));
typedef float    float8  __attribute__((ext_vector_type(8)));

#define WMMA_F16(a, b, c) \
  __builtin_amdgcn_wmma_f32_16x16x32_f16(false, (a), false, (b), (short)0, (c), false, false)

__device__ __forceinline__ half16 mkfrag(half8 lo, half8 hi) {
  half16 f;
#pragma unroll
  for (int e = 0; e < 8; ++e) { f[e] = lo[e]; f[e + 8] = hi[e]; }
  return f;
}
__device__ __forceinline__ half8 ld8(const _Float16* p) { return *(const half8*)p; }

// ---------------------------------------------------------------------------
// CDNA5 async global->LDS (ASYNCcnt-tracked), per cdna5_isa 15.18.3 / 08.
// VDST = per-lane LDS byte address, VADDR = per-lane 64-bit global address.
// ---------------------------------------------------------------------------
__device__ __forceinline__ uint32_t lds_off(const void* p) {
  return (uint32_t)(uintptr_t)(const __attribute__((address_space(3))) void*)p;
}
__device__ __forceinline__ void async_g2l_b128(uint32_t ldsaddr, const void* gaddr) {
  asm volatile("global_load_async_to_lds_b128 %0, %1, off"
               :: "v"(ldsaddr), "v"(gaddr) : "memory");
}
__device__ __forceinline__ void wait_async_le5() {
  asm volatile("s_wait_asynccnt 5" ::: "memory");
}
__device__ __forceinline__ void wait_async_0() {
  asm volatile("s_wait_asynccnt 0" ::: "memory");
}

// ---------------------------------------------------------------------------
// Problem constants
// ---------------------------------------------------------------------------
#define BQ      2
#define NSEQ    2048
#define DIMM    1024
#define NHEADS  16
#define DHEAD   64
#define WIN     256
#define NW      (NSEQ / WIN)          // 8
#define MROWS   (BQ * NSEQ)           // 4096
#define NQKV    (3 * DIMM)            // 3072
#define DINNER  2730
#define NH1P    5632                  // 2*DINNER=5460 padded to 22*256
#define KP2     2752                  // DINNER=2730 padded to 86*32
#define DEPTH   6

// ---------------------------------------------------------------------------
// GEMM: C[M,N] = A[M,K](f16) * Bt[N,K](f16)^T (+bias)(+resid), out f16 or f32
// Block 256 thr = 8 waves (2M x 4N); wave tile 32x64; block tile 64x256.
// K-step 32 tiles double-buffered in LDS via async global->LDS b128.
// ---------------------------------------------------------------------------
#define GA_ELEMS (64 * 32)            // A tile
#define GB_ELEMS (256 * 32)           // B tile

__device__ __forceinline__ void gemm_stage(
    const _Float16* __restrict__ A, const _Float16* __restrict__ Bt,
    _Float16* ash, _Float16* bsh, int m0g, int n0g, int K, int k0,
    int wid, int lane)
{
  const int lr = lane >> 2;           // 0..7  (row within group of 8)
  const int kc = lane & 3;            // 0..3  (16B chunk along K)
  {
    const int row = wid * 8 + lr;     // 64 A rows across 8 waves
    async_g2l_b128(lds_off(ash + row * 32 + kc * 8),
                   (const void*)(A + (size_t)(m0g + row) * K + k0 + kc * 8));
  }
#pragma unroll
  for (int it = 0; it < 4; ++it) {    // 256 B rows across 8 waves
    const int row = wid * 32 + it * 8 + lr;
    async_g2l_b128(lds_off(bsh + row * 32 + kc * 8),
                   (const void*)(Bt + (size_t)(n0g + row) * K + k0 + kc * 8));
  }
}

template <bool OUT_F32>
__global__ __launch_bounds__(256) void gemm_f16_kernel(
    const _Float16* __restrict__ A, const _Float16* __restrict__ Bt,
    const float* __restrict__ bias, int biasN,
    const float* __restrict__ resid, void* __restrict__ outp,
    int M, int N, int K)
{
  __shared__ _Float16 Ash[2][GA_ELEMS];
  __shared__ _Float16 Bsh[2][GB_ELEMS];

  const int lane   = threadIdx.x & 31;
  const int wid    = threadIdx.x >> 5;
  const int laneHi = (lane >> 4) & 1;
  const int ln     = lane & 15;
  const int wm = wid & 1, wn = wid >> 1;      // 2 x 4 wave grid
  const int m0g = blockIdx.y * 64;
  const int n0g = blockIdx.x * 256;

  float8 acc[2][4] = {};

  const int nsteps = K / 32;
  gemm_stage(A, Bt, Ash[0], Bsh[0], m0g, n0g, K, 0, wid, lane);

  for (int i = 0; i < nsteps; ++i) {
    const int cur = i & 1;
    const bool haveNext = (i + 1) < nsteps;
    if (haveNext)
      gemm_stage(A, Bt, Ash[cur ^ 1], Bsh[cur ^ 1], m0g, n0g, K, (i + 1) * 32, wid, lane);
    if (haveNext) wait_async_le5(); else wait_async_0();
    __syncthreads();

    const _Float16* ab = Ash[cur];
    const _Float16* bb = Bsh[cur];
    const _Float16* ar0 = ab + (wm * 32 + ln) * 32 + laneHi * 8;
    const _Float16* ar1 = ab + (wm * 32 + 16 + ln) * 32 + laneHi * 8;
    half16 a0 = mkfrag(ld8(ar0), ld8(ar0 + 16));
    half16 a1 = mkfrag(ld8(ar1), ld8(ar1 + 16));
#pragma unroll
    for (int fg = 0; fg < 4; ++fg) {
      const _Float16* br = bb + (wn * 64 + fg * 16 + ln) * 32 + laneHi * 8;
      half16 bfr = mkfrag(ld8(br), ld8(br + 16));
      acc[0][fg] = WMMA_F16(a0, bfr, acc[0][fg]);
      acc[1][fg] = WMMA_F16(a1, bfr, acc[1][fg]);
    }
    __syncthreads();
  }

#pragma unroll
  for (int mf = 0; mf < 2; ++mf) {
#pragma unroll
    for (int fg = 0; fg < 4; ++fg) {
#pragma unroll
      for (int r = 0; r < 8; ++r) {
        const int row = m0g + wm * 32 + mf * 16 + laneHi * 8 + r;
        const int col = n0g + wn * 64 + fg * 16 + ln;
        float v = acc[mf][fg][r];
        if (bias && col < biasN) v += bias[col];
        if (resid) v += resid[(size_t)row * N + col];
        if (OUT_F32)
          ((float*)outp)[(size_t)row * N + col] = v;
        else
          ((_Float16*)outp)[(size_t)row * N + col] = (_Float16)v;
      }
    }
  }
}

// ---------------------------------------------------------------------------
// Weight convert + transpose + pad:  in f32 [K,N] -> out f16 [Np,Kp]
// ---------------------------------------------------------------------------
__global__ void transpose_w_kernel(const float* __restrict__ in,
                                   _Float16* __restrict__ out,
                                   int K, int N, int Kp, int Np)
{
  size_t idx = (size_t)blockIdx.x * 256 + threadIdx.x;
  if (idx >= (size_t)Np * Kp) return;
  int k = (int)(idx % Kp);
  int n = (int)(idx / Kp);
  float v = (n < N && k < K) ? in[(size_t)k * N + n] : 0.f;
  out[idx] = (_Float16)v;
}

// ---------------------------------------------------------------------------
// shift_tokens (attn input): f32 x -> f16 xs
// ---------------------------------------------------------------------------
__global__ void shiftA_kernel(const float* __restrict__ x, _Float16* __restrict__ xs)
{
  size_t idx = (size_t)blockIdx.x * 256 + threadIdx.x;  // MROWS*DIMM
  int col = (int)(idx & (DIMM - 1));
  size_t row = idx >> 10;
  int n = (int)(row & (NSEQ - 1));
  float v;
  if (col < DIMM / 2) v = x[idx];
  else                v = (n > 0) ? x[idx - DIMM] : 0.f;
  xs[idx] = (_Float16)v;
}

// ---------------------------------------------------------------------------
// shift_tokens + layernorm (FFN input): one block (256 thr) per row
// ---------------------------------------------------------------------------
__global__ __launch_bounds__(256) void shiftLN_kernel(
    const float* __restrict__ x, const float* __restrict__ g,
    const float* __restrict__ bt, _Float16* __restrict__ xn)
{
  __shared__ float sred[8], sred2[8], stat[2];
  const int row = blockIdx.x;
  const int n = row & (NSEQ - 1);
  const float* xr = x + (size_t)row * DIMM;
  const float* xp = xr - DIMM;

  float vals[4]; float s = 0.f, s2 = 0.f;
#pragma unroll
  for (int u = 0; u < 4; ++u) {
    int col = threadIdx.x + u * 256;
    float v = (col < DIMM / 2) ? xr[col] : (n > 0 ? xp[col] : 0.f);
    vals[u] = v; s += v; s2 += v * v;
  }
  for (int off = 16; off >= 1; off >>= 1) {
    s  += __shfl_xor(s, off, 32);
    s2 += __shfl_xor(s2, off, 32);
  }
  int lane = threadIdx.x & 31, wv = threadIdx.x >> 5;
  if (lane == 0) { sred[wv] = s; sred2[wv] = s2; }
  __syncthreads();
  if (threadIdx.x == 0) {
    float a = 0.f, c = 0.f;
    for (int i = 0; i < 8; ++i) { a += sred[i]; c += sred2[i]; }
    float mu = a / (float)DIMM;
    float var = c / (float)DIMM - mu * mu;
    stat[0] = mu; stat[1] = rsqrtf(var + 1e-5f);
  }
  __syncthreads();
  float mu = stat[0], is = stat[1];
#pragma unroll
  for (int u = 0; u < 4; ++u) {
    int col = threadIdx.x + u * 256;
    xn[(size_t)row * DIMM + col] = (_Float16)(((vals[u] - mu) * is) * g[col] + bt[col]);
  }
}

// ---------------------------------------------------------------------------
// QKV post-process: l2norm * scale, xpos-rotary.  One wave per (b,h,t),
// each lane owns d=lane and d=lane+32 (rotate_half partners).
// ---------------------------------------------------------------------------
__global__ __launch_bounds__(256) void qkv_process_kernel(
    const _Float16* __restrict__ QKV, const float* __restrict__ qs,
    const float* __restrict__ ks, _Float16* __restrict__ Qo,
    _Float16* __restrict__ Ko, _Float16* __restrict__ Vo)
{
  const int lane = threadIdx.x & 31;
  const int wv   = threadIdx.x >> 5;
  const int widx = blockIdx.x * 8 + wv;                  // b*H*N + h*N + t
  const int b = widx >> 15;
  const int rem = widx & 32767;
  const int h = rem >> 11;
  const int t = rem & (NSEQ - 1);

  const _Float16* rowp = QKV + ((size_t)b * NSEQ + t) * NQKV + h * DHEAD;
  const int d0 = lane, d1 = lane + 32;

  float q0 = rowp[d0],            q1 = rowp[d1];
  float k0 = rowp[DIMM + d0],     k1 = rowp[DIMM + d1];
  float v0 = rowp[2 * DIMM + d0], v1 = rowp[2 * DIMM + d1];

  float sq = q0 * q0 + q1 * q1;
  float sk = k0 * k0 + k1 * k1;
  for (int off = 16; off >= 1; off >>= 1) {
    sq += __shfl_xor(sq, off, 32);
    sk += __shfl_xor(sk, off, 32);
  }
  float qi = 1.f / fmaxf(sqrtf(sq), 1e-12f);
  float ki = 1.f / fmaxf(sqrtf(sk), 1e-12f);
  q0 *= qi * qs[d0]; q1 *= qi * qs[d1];
  k0 *= ki * ks[d0]; k1 *= ki * ks[d1];

  // rotary + xpos
  float invf = powf(10000.f, -(2.f * (float)lane) / 64.f);
  float fr = (float)t * invf;
  float c = cosf(fr), s = sinf(fr);
  float base = (2.f * (float)lane + 0.4f * 64.f) / (1.4f * 64.f);
  float scl  = powf(base, ((float)t - (float)(NSEQ / 2)) / (float)(WIN / 2));
  float iscl = 1.f / scl;

  float qo0 = (q0 * c - q1 * s) * scl;
  float qo1 = (q1 * c + q0 * s) * scl;
  float ko0 = (k0 * c - k1 * s) * iscl;
  float ko1 = (k1 * c + k0 * s) * iscl;

  const size_t ob = ((size_t)b * NHEADS + h) * NSEQ + t;
  Qo[ob * DHEAD + d0] = (_Float16)qo0; Qo[ob * DHEAD + d1] = (_Float16)qo1;
  Ko[ob * DHEAD + d0] = (_Float16)ko0; Ko[ob * DHEAD + d1] = (_Float16)ko1;
  Vo[ob * DHEAD + d0] = (_Float16)v0;  Vo[ob * DHEAD + d1] = (_Float16)v1;
}

// ---------------------------------------------------------------------------
// Local windowed attention, flash-style online softmax.
// Block = 128 threads (4 waves), each wave owns 16 query rows.
// grid = B*H*NW*2 = 512 blocks.  Keys = prev window + cur window (2 phases).
// ---------------------------------------------------------------------------
#define VT_PITCH 264
__global__ __launch_bounds__(128) void attn_kernel(
    const _Float16* __restrict__ Q, const _Float16* __restrict__ Kk,
    const _Float16* __restrict__ V, _Float16* __restrict__ O)
{
  __shared__ _Float16 vtsh[DHEAD][VT_PITCH];   // V^T for current 256-key phase
  __shared__ _Float16 psh[4][16 * 32];         // per-wave prob staging (D->A relayout)

  const int tid = threadIdx.x;
  const int lane = tid & 31, wv = tid >> 5;
  const int laneHi = (lane >> 4) & 1;
  const int ln = lane & 15;

  const int blk = blockIdx.x;
  const int qh = blk & 1;
  const int w  = (blk >> 1) & 7;
  const int h  = (blk >> 4) & 15;
  const int b  = blk >> 8;

  const size_t bh = ((size_t)b * NHEADS + h) * NSEQ;
  const int qbase = w * WIN + qh * 64 + wv * 16;

  // Q fragments (held in registers for the whole kernel)
  const _Float16* qrow = Q + (bh + qbase + ln) * DHEAD;
  half16 aq0 = mkfrag(ld8(qrow + laneHi * 8),      ld8(qrow + laneHi * 8 + 16));
  half16 aq1 = mkfrag(ld8(qrow + 32 + laneHi * 8), ld8(qrow + 32 + laneHi * 8 + 16));

  float mrow[8], lrow[8];
#pragma unroll
  for (int r = 0; r < 8; ++r) { mrow[r] = -1e30f; lrow[r] = 0.f; }
  float8 oacc[4] = {};

  for (int phase = 0; phase < 2; ++phase) {
    if (w == 0 && phase == 0) continue;                 // padded block: all masked
    const int jstart = w * WIN - WIN + phase * WIN;     // global token of local j=0

    __syncthreads();
    for (int idx = tid; idx < WIN * DHEAD; idx += 128) {
      int jl = idx >> 6, d = idx & 63;
      int jg = jstart + jl;
      vtsh[d][jl] = V[(bh + jg) * DHEAD + d];           // jg >= 0 whenever executed
    }
    __syncthreads();

    for (int kk = 0; kk < 8; ++kk) {            // 8 K-steps of 32 keys
#pragma unroll
      for (int sub = 0; sub < 2; ++sub) {       // 16-key score chunks
        const int jl = kk * 32 + sub * 16 + ln;
        const int jg = jstart + jl;
        const _Float16* krow = Kk + (bh + jg) * DHEAD;
        half16 bk0 = mkfrag(ld8(krow + laneHi * 8),      ld8(krow + laneHi * 8 + 16));
        half16 bk1 = mkfrag(ld8(krow + 32 + laneHi * 8), ld8(krow + 32 + laneHi * 8 + 16));
        float8 cfr = {};
        cfr = WMMA_F16(aq0, bk0, cfr);
        cfr = WMMA_F16(aq1, bk1, cfr);

        float sv[8], rm[8];
        bool  msk[8];
#pragma unroll
        for (int r = 0; r < 8; ++r) {
          int i = qbase + laneHi * 8 + r;
          msk[r] = (i < jg) || (i > jg + WIN);
          sv[r]  = msk[r] ? -1e30f : cfr[r] * 8.0f;     // QK_SCALE
          rm[r]  = sv[r];
        }
#pragma unroll
        for (int r = 0; r < 8; ++r)
          for (int off = 1; off < 16; off <<= 1)
            rm[r] = fmaxf(rm[r], __shfl_xor(rm[r], off, 32));

        float pv[8];
#pragma unroll
        for (int r = 0; r < 8; ++r) {
          float mnew = fmaxf(mrow[r], rm[r]);
          float fac  = __expf(mrow[r] - mnew);
          lrow[r] *= fac;
#pragma unroll
          for (int fg = 0; fg < 4; ++fg) oacc[fg][r] *= fac;
          pv[r] = msk[r] ? 0.f : __expf(sv[r] - mnew);
          mrow[r] = mnew;
        }
        float ps[8];
#pragma unroll
        for (int r = 0; r < 8; ++r) ps[r] = pv[r];
#pragma unroll
        for (int r = 0; r < 8; ++r)
          for (int off = 1; off < 16; off <<= 1)
            ps[r] += __shfl_xor(ps[r], off, 32);
#pragma unroll
        for (int r = 0; r < 8; ++r) {
          lrow[r] += ps[r];
          psh[wv][(laneHi * 8 + r) * 32 + sub * 16 + ln] = (_Float16)pv[r];
        }
      }

      // P (16x32) x V (32x64) via WMMA, P re-read from LDS in A layout
      const _Float16* pr = &psh[wv][0] + ln * 32;
      half16 ap = mkfrag(ld8(pr + laneHi * 8), ld8(pr + laneHi * 8 + 16));
#pragma unroll
      for (int fg = 0; fg < 4; ++fg) {
        const _Float16* vr = &vtsh[fg * 16 + ln][kk * 32 + laneHi * 8];
        half16 bv = mkfrag(ld8(vr), ld8(vr + 16));
        oacc[fg] = WMMA_F16(ap, bv, oacc[fg]);
      }
    }
  }

  // write out [B*N, INNER] with col = h*64 + d
#pragma unroll
  for (int fg = 0; fg < 4; ++fg) {
#pragma unroll
    for (int r = 0; r < 8; ++r) {
      int i   = qbase + laneHi * 8 + r;
      int col = h * DHEAD + fg * 16 + ln;
      float ov = oacc[fg][r] / lrow[r];
      O[((size_t)b * NSEQ + i) * DIMM + col] = (_Float16)ov;
    }
  }
}

// ---------------------------------------------------------------------------
// GEGLU: y[:,c] = a * gelu_exact(gate), c<DINNER; padded cols -> 0
// ---------------------------------------------------------------------------
__global__ void geglu_kernel(const _Float16* __restrict__ hg, _Float16* __restrict__ y)
{
  size_t idx = (size_t)blockIdx.x * 256 + threadIdx.x;   // MROWS*KP2
  int col = (int)(idx % KP2);
  size_t row = idx / KP2;
  float o = 0.f;
  if (col < DINNER) {
    float a = hg[row * NH1P + col];
    float g = hg[row * NH1P + DINNER + col];
    o = a * 0.5f * g * (1.f + erff(g * 0.70710678f));
  }
  y[row * KP2 + col] = (_Float16)o;
}

// ---------------------------------------------------------------------------
// Host launcher
// ---------------------------------------------------------------------------
static inline size_t alignup(size_t v) { return (v + 255) & ~(size_t)255; }

extern "C" void kernel_launch(void* const* d_in, const int* in_sizes, int n_in,
                              void* d_out, int out_size, void* d_ws, size_t ws_size,
                              hipStream_t stream) {
  const float* x_in  = (const float*)d_in[0];
  const float* Wqkv  = (const float*)d_in[1];
  const float* qscal = (const float*)d_in[2];
  const float* kscal = (const float*)d_in[3];
  const float* Wo    = (const float*)d_in[4];
  const float* ln_g  = (const float*)d_in[5];
  const float* ln_b  = (const float*)d_in[6];
  const float* W1    = (const float*)d_in[7];
  const float* b1    = (const float*)d_in[8];
  const float* W2    = (const float*)d_in[9];
  const float* b2    = (const float*)d_in[10];
  float* x = (float*)d_out;                         // running activation (f32)

  // workspace carve-up (all f16)
  char* wsb = (char*)d_ws;
  size_t off = 0;
  auto alloc = [&](size_t elems) -> _Float16* {
    _Float16* p = (_Float16*)(wsb + off);
    off = alignup(off + elems * sizeof(_Float16));
    return p;
  };
  _Float16* xs_h   = alloc((size_t)MROWS * DIMM);
  _Float16* qkv_h  = alloc((size_t)MROWS * NQKV);
  _Float16* q_p    = alloc((size_t)MROWS * DIMM);
  _Float16* k_p    = alloc((size_t)MROWS * DIMM);
  _Float16* v_p    = alloc((size_t)MROWS * DIMM);
  _Float16* attn_h = alloc((size_t)MROWS * DIMM);
  _Float16* xn_h   = alloc((size_t)MROWS * DIMM);
  _Float16* hg_h   = alloc((size_t)MROWS * NH1P);
  _Float16* y_h    = alloc((size_t)MROWS * KP2);
  _Float16* wqkv_t = alloc((size_t)NQKV * DIMM);
  _Float16* wo_t   = alloc((size_t)DIMM * DIMM);
  _Float16* w1_t   = alloc((size_t)NH1P * DIMM);
  _Float16* w2_t   = alloc((size_t)DIMM * KP2);

  // x = input
  hipMemcpyAsync(x, x_in, (size_t)MROWS * DIMM * sizeof(float),
                 hipMemcpyDeviceToDevice, stream);

  for (int L = 0; L < DEPTH; ++L) {
    const float* lWqkv = Wqkv + (size_t)L * DIMM * NQKV;
    const float* lWo   = Wo   + (size_t)L * DIMM * DIMM;
    const float* lW1   = W1   + (size_t)L * DIMM * (2 * DINNER);
    const float* lb1   = b1   + (size_t)L * (2 * DINNER);
    const float* lW2   = W2   + (size_t)L * DINNER * DIMM;
    const float* lb2   = b2   + (size_t)L * DIMM;
    const float* lg    = ln_g + (size_t)L * DIMM;
    const float* lbn   = ln_b + (size_t)L * DIMM;
    const float* lqs   = qscal + (size_t)L * DHEAD;
    const float* lks   = kscal + (size_t)L * DHEAD;

    // ---- weight prep (convert + transpose + pad) ----
    transpose_w_kernel<<<(NQKV * DIMM) / 256, 256, 0, stream>>>(lWqkv, wqkv_t, DIMM, NQKV, DIMM, NQKV);
    transpose_w_kernel<<<(DIMM * DIMM) / 256, 256, 0, stream>>>(lWo, wo_t, DIMM, DIMM, DIMM, DIMM);
    transpose_w_kernel<<<((size_t)NH1P * DIMM) / 256, 256, 0, stream>>>(lW1, w1_t, DIMM, 2 * DINNER, DIMM, NH1P);
    transpose_w_kernel<<<((size_t)DIMM * KP2) / 256, 256, 0, stream>>>(lW2, w2_t, DINNER, DIMM, KP2, DIMM);

    // ---- attention half ----
    shiftA_kernel<<<(MROWS * DIMM) / 256, 256, 0, stream>>>(x, xs_h);
    gemm_f16_kernel<false><<<dim3(NQKV / 256, MROWS / 64), 256, 0, stream>>>(
        xs_h, wqkv_t, nullptr, 0, nullptr, qkv_h, MROWS, NQKV, DIMM);
    qkv_process_kernel<<<(BQ * NHEADS * NSEQ) / 8, 256, 0, stream>>>(
        qkv_h, lqs, lks, q_p, k_p, v_p);
    attn_kernel<<<BQ * NHEADS * NW * 2, 128, 0, stream>>>(q_p, k_p, v_p, attn_h);
    gemm_f16_kernel<true><<<dim3(DIMM / 256, MROWS / 64), 256, 0, stream>>>(
        attn_h, wo_t, nullptr, 0, x, (void*)x, MROWS, DIMM, DIMM);

    // ---- FFN half ----
    shiftLN_kernel<<<MROWS, 256, 0, stream>>>(x, lg, lbn, xn_h);
    gemm_f16_kernel<false><<<dim3(NH1P / 256, MROWS / 64), 256, 0, stream>>>(
        xn_h, w1_t, lb1, 2 * DINNER, nullptr, hg_h, MROWS, NH1P, DIMM);
    geglu_kernel<<<((size_t)MROWS * KP2) / 256, 256, 0, stream>>>(hg_h, y_h);
    gemm_f16_kernel<true><<<dim3(DIMM / 256, MROWS / 64), 256, 0, stream>>>(
        y_h, w2_t, lb2, DIMM, x, (void*)x, MROWS, DIMM, KP2);
  }
  (void)in_sizes; (void)n_in; (void)out_size; (void)ws_size;
}
